// BERTRec_sequential_41291815584150
// MI455X (gfx1250) — matrix operand order repaired and checked by reading, not compiled, not run on hardware
//
#include <hip/hip_runtime.h>
#include <hip/hip_bf16.h>
#include <math.h>

// ---------------- model constants ----------------
#define BB   32
#define LLEN 200
#define DD   256
#define HH   4
#define DKK  64
#define FFD  1024
#define NTOK (BB * LLEN)      // 6400
#define NOUT 30001            // ITEM + 1

typedef __bf16 bf16_t;
typedef __attribute__((ext_vector_type(16))) __bf16 v16bf;
typedef __attribute__((ext_vector_type(8)))  __bf16 v8bf;
typedef __attribute__((ext_vector_type(8)))  float  v8f;

__device__ __forceinline__ bf16_t to_bf(float f) { return (bf16_t)f; }

// ---------------- small utility kernels ----------------
__global__ void zero_f32_kernel(float* __restrict__ p, int n) {
    int i = blockIdx.x * blockDim.x + threadIdx.x;
    if (i < n) p[i] = 0.0f;
}

__global__ void f32_to_bf16_kernel(const float* __restrict__ src,
                                   bf16_t* __restrict__ dst, int n) {
    int i = blockIdx.x * blockDim.x + threadIdx.x;
    if (i < n) dst[i] = to_bf(src[i]);
}

// Embedding gather: dst[n, e] = bf16(table[x[n], e]); one block per token row.
__global__ void gather_bf16_kernel(const long long* __restrict__ x,
                                   const float* __restrict__ table,
                                   bf16_t* __restrict__ dst, int E) {
    int n = blockIdx.x;
    long long row = x[n];
    const float* src = table + (long long)row * E;
    bf16_t* d = dst + (long long)n * E;
    for (int e = threadIdx.x; e < E; e += blockDim.x) d[e] = to_bf(src[e]);
}

// ---------------- WMMA bf16 GEMM: C = act(A[M,K] @ W[N,K]^T + bias) ----------------
// Wave tile: 32 rows x 64 cols = 2x4 WMMA accumulators.
// A-frag (16x32 bf16): lane holds row m = lane&15; K chunks at half*8 and 16+half*8.
// B-frag (32x16 bf16): lane holds col n = lane&15 (= row n of W since B = W^T);
//                      K chunks at half*8 and 16+half*8 (same order as A).
// Out-of-range B rows are CLAMPED to N-1 (not masked): the garbage columns are
// never stored, and clamping keeps EXEC all-ones through the whole K loop
// (required for WMMA, and avoids per-iteration saveexec divergence).

__device__ __forceinline__ v16bf frag_a_ld(const bf16_t* __restrict__ p) {
    v8bf c0 = *(const v8bf*)(p);
    v8bf c1 = *(const v8bf*)(p + 16);
    v16bf f;
#pragma unroll
    for (int i = 0; i < 8; ++i) { f[i] = c0[i]; f[i + 8] = c1[i]; }
    return f;
}

__device__ __forceinline__ v16bf frag_b_ld(const bf16_t* __restrict__ p) {
    v8bf c0 = *(const v8bf*)(p);
    v8bf c1 = *(const v8bf*)(p + 16);
    v16bf f;
#pragma unroll
    for (int i = 0; i < 8; ++i) { f[i] = c0[i]; f[i + 8] = c1[i]; }
    return f;
}

#define ACT_NONE  0
#define ACT_LEAKY 1
#define ACT_GELU  2

template <int ACT>
__global__ void __launch_bounds__(256)
gemm_bf16_kernel(const bf16_t* __restrict__ A,   // [M,K]
                 const bf16_t* __restrict__ W,   // [N,K]
                 const float*  __restrict__ bias,// [N] or null
                 float*        __restrict__ Cf,  // [M,N] or null
                 bf16_t*       __restrict__ Cb,  // [M,N] or null
                 int M, int N, int K) {
    const int lane = threadIdx.x & 31;
    const int half = lane >> 4;
    const int r    = lane & 15;
    const int tiles_n = (N + 63) >> 6;
    const int tiles_m = (M + 31) >> 5;
    const int wid = blockIdx.x * (blockDim.x >> 5) + (threadIdx.x >> 5);
    if (wid >= tiles_m * tiles_n) return;   // wave-uniform: EXEC stays all-ones
    const int tm = wid / tiles_n;
    const int tn = wid % tiles_n;
    const int row0 = tm << 5;
    const int col0 = tn << 6;

    v8f acc[2][4];
#pragma unroll
    for (int i = 0; i < 2; ++i)
#pragma unroll
        for (int j = 0; j < 4; ++j)
#pragma unroll
            for (int e = 0; e < 8; ++e) acc[i][j][e] = 0.0f;

    // K-invariant per-lane base pointers (hoisted out of the hot loop).
    const bf16_t* pa[2];
#pragma unroll
    for (int i = 0; i < 2; ++i)
        pa[i] = A + (size_t)(row0 + i * 16 + r) * K + half * 8;
    const bf16_t* pb[4];
#pragma unroll
    for (int j = 0; j < 4; ++j) {
        int br = col0 + j * 16 + r;
        if (br > N - 1) br = N - 1;         // clamp: keeps loads valid, no divergence
        pb[j] = W + (size_t)br * K + half * 8;
    }

    for (int k0 = 0; k0 < K; k0 += 32) {
        v16bf a0 = frag_a_ld(pa[0] + k0);
        v16bf a1 = frag_a_ld(pa[1] + k0);
        v16bf b0 = frag_b_ld(pb[0] + k0);
        v16bf b1 = frag_b_ld(pb[1] + k0);
        v16bf b2 = frag_b_ld(pb[2] + k0);
        v16bf b3 = frag_b_ld(pb[3] + k0);
        acc[0][0] = __builtin_amdgcn_wmma_f32_16x16x32_bf16(false, a0, false, b0,
                        (short)0, acc[0][0], false, false);
        acc[0][1] = __builtin_amdgcn_wmma_f32_16x16x32_bf16(false, a0, false, b1,
                        (short)0, acc[0][1], false, false);
        acc[0][2] = __builtin_amdgcn_wmma_f32_16x16x32_bf16(false, a0, false, b2,
                        (short)0, acc[0][2], false, false);
        acc[0][3] = __builtin_amdgcn_wmma_f32_16x16x32_bf16(false, a0, false, b3,
                        (short)0, acc[0][3], false, false);
        acc[1][0] = __builtin_amdgcn_wmma_f32_16x16x32_bf16(false, a1, false, b0,
                        (short)0, acc[1][0], false, false);
        acc[1][1] = __builtin_amdgcn_wmma_f32_16x16x32_bf16(false, a1, false, b1,
                        (short)0, acc[1][1], false, false);
        acc[1][2] = __builtin_amdgcn_wmma_f32_16x16x32_bf16(false, a1, false, b2,
                        (short)0, acc[1][2], false, false);
        acc[1][3] = __builtin_amdgcn_wmma_f32_16x16x32_bf16(false, a1, false, b3,
                        (short)0, acc[1][3], false, false);
    }

#pragma unroll
    for (int mi = 0; mi < 2; ++mi)
#pragma unroll
        for (int ni = 0; ni < 4; ++ni) {
            int col = col0 + ni * 16 + r;
            if (col >= N) continue;
            float bs = bias ? bias[col] : 0.0f;
#pragma unroll
            for (int e = 0; e < 8; ++e) {
                int rowg = row0 + mi * 16 + half * 8 + e;
                if (rowg >= M) continue;
                float v = acc[mi][ni][e] + bs;
                if (ACT == ACT_LEAKY) {
                    v = v >= 0.0f ? v : 0.01f * v;
                } else if (ACT == ACT_GELU) {
                    float xx = v;
                    v = 0.5f * xx * (1.0f + tanhf(0.7978845608028654f *
                                                  (xx + 0.044715f * xx * xx * xx)));
                }
                size_t idx = (size_t)rowg * N + col;
                if (Cf) Cf[idx] = v;
                if (Cb) Cb[idx] = to_bf(v);
            }
        }
}

// ---------------- LayerNorm kernels (one 256-thread block per token row) ----------------
// h = std_ln(hin + pos); writes f32 + bf16.
__global__ void ln0_kernel(const float* __restrict__ hin, const float* __restrict__ pos,
                           const float* __restrict__ w, const float* __restrict__ b,
                           float* __restrict__ hout, bf16_t* __restrict__ hbf) {
    __shared__ float red[DD];
    int n = blockIdx.x, d = threadIdx.x;
    size_t idx = (size_t)n * DD + d;
    float val = hin[idx] + pos[(n % LLEN) * DD + d];
    red[d] = val; __syncthreads();
    for (int off = DD / 2; off > 0; off >>= 1) {
        if (d < off) red[d] += red[d + off];
        __syncthreads();
    }
    float mean = red[0] * (1.0f / DD); __syncthreads();
    float c = val - mean;
    red[d] = c * c; __syncthreads();
    for (int off = DD / 2; off > 0; off >>= 1) {
        if (d < off) red[d] += red[d + off];
        __syncthreads();
    }
    float var = red[0] * (1.0f / DD);
    float o = c * rsqrtf(var + 1e-5f) * w[d] + b[d];
    hout[idx] = o;
    hbf[idx] = to_bf(o);
}

// h = custom_ln(h + delta): unbiased std (ddof=1), eps added to std.
__global__ void cln_res_kernel(float* __restrict__ hio, const float* __restrict__ delta,
                               const float* __restrict__ a, const float* __restrict__ c,
                               bf16_t* __restrict__ hbf) {
    __shared__ float red[DD];
    int n = blockIdx.x, d = threadIdx.x;
    size_t idx = (size_t)n * DD + d;
    float val = hio[idx] + delta[idx];
    red[d] = val; __syncthreads();
    for (int off = DD / 2; off > 0; off >>= 1) {
        if (d < off) red[d] += red[d + off];
        __syncthreads();
    }
    float mean = red[0] * (1.0f / DD); __syncthreads();
    float ct = val - mean;
    red[d] = ct * ct; __syncthreads();
    for (int off = DD / 2; off > 0; off >>= 1) {
        if (d < off) red[d] += red[d + off];
        __syncthreads();
    }
    float s = sqrtf(red[0] * (1.0f / (DD - 1)));  // ddof=1
    float o = a[d] * ct / (s + 1e-6f) + c[d];
    hio[idx] = o;
    hbf[idx] = to_bf(o);
}

// y += std_ln(h, w, b)
__global__ void topln_accum_kernel(const float* __restrict__ h,
                                   const float* __restrict__ w, const float* __restrict__ b,
                                   float* __restrict__ y) {
    __shared__ float red[DD];
    int n = blockIdx.x, d = threadIdx.x;
    size_t idx = (size_t)n * DD + d;
    float val = h[idx];
    red[d] = val; __syncthreads();
    for (int off = DD / 2; off > 0; off >>= 1) {
        if (d < off) red[d] += red[d + off];
        __syncthreads();
    }
    float mean = red[0] * (1.0f / DD); __syncthreads();
    float c = val - mean;
    red[d] = c * c; __syncthreads();
    for (int off = DD / 2; off > 0; off >>= 1) {
        if (d < off) red[d] += red[d + off];
        __syncthreads();
    }
    float var = red[0] * (1.0f / DD);
    y[idx] += c * rsqrtf(var + 1e-5f) * w[d] + b[d];
}

// ---------------- fused attention: scores + mask + softmax + A@V ----------------
// One block per (b, h, i). q/k/v are [B*L, D] with head-interleaved D (= h*DK + d).
__global__ void attn_kernel(const float* __restrict__ q, const float* __restrict__ k,
                            const float* __restrict__ v, const long long* __restrict__ x,
                            bf16_t* __restrict__ obf) {
    __shared__ float s[LLEN];
    __shared__ float red[256];
    int bi = blockIdx.x;
    int i  = bi % LLEN;
    int bh = bi / LLEN;
    int h  = bh % HH;
    int b  = bh / HH;
    int t  = threadIdx.x;

    const float* qrow = q + ((size_t)(b * LLEN + i) * DD + h * DKK);
    for (int j = t; j < LLEN; j += blockDim.x) {
        const float* krow = k + ((size_t)(b * LLEN + j) * DD + h * DKK);
        float dot = 0.0f;
#pragma unroll 8
        for (int d = 0; d < DKK; ++d) dot += qrow[d] * krow[d];
        s[j] = (x[b * LLEN + j] > 0) ? dot * 0.125f : -1e9f;  // 1/sqrt(64)
    }
    __syncthreads();

    float mx = -INFINITY;
    for (int j = t; j < LLEN; j += blockDim.x) mx = fmaxf(mx, s[j]);
    red[t] = mx; __syncthreads();
    for (int off = 128; off > 0; off >>= 1) {
        if (t < off) red[t] = fmaxf(red[t], red[t + off]);
        __syncthreads();
    }
    float rowmax = red[0]; __syncthreads();

    float sm = 0.0f;
    for (int j = t; j < LLEN; j += blockDim.x) {
        float e = __expf(s[j] - rowmax);
        s[j] = e;
        sm += e;
    }
    red[t] = sm; __syncthreads();
    for (int off = 128; off > 0; off >>= 1) {
        if (t < off) red[t] += red[t + off];
        __syncthreads();
    }
    float inv = 1.0f / red[0];

    if (t < DKK) {
        float o = 0.0f;
        for (int j = 0; j < LLEN; ++j)
            o += s[j] * v[(size_t)(b * LLEN + j) * DD + h * DKK + t];
        obf[(size_t)(b * LLEN + i) * DD + h * DKK + t] = to_bf(o * inv);
    }
}

// ---------------- host orchestration ----------------
extern "C" void kernel_launch(void* const* d_in, const int* in_sizes, int n_in,
                              void* d_out, int out_size, void* d_ws, size_t ws_size,
                              hipStream_t stream) {
    (void)in_sizes; (void)n_in; (void)out_size; (void)ws_size;

    static const int EMB[4] = {768, 128, 512, 128};
    static const int LHS[4] = {512, 320, 384, 224};

    const long long* x = (const long long*)d_in[0];
    const float* out_w = (const float*)d_in[1 + 4 * 42];
    const float* out_b = (const float*)d_in[1 + 4 * 42 + 1];

    // -------- carve workspace --------
    size_t off = 0;
    auto carve = [&](size_t bytes) -> char* {
        char* p = (char*)d_ws + off;
        off += (bytes + 255) & ~(size_t)255;
        return p;
    };
    float*  y    = (float*) carve((size_t)NTOK * DD * 4);
    float*  h    = (float*) carve((size_t)NTOK * DD * 4);
    bf16_t* hbf  = (bf16_t*)carve((size_t)NTOK * DD * 2);
    float*  qf   = (float*) carve((size_t)NTOK * DD * 4);
    float*  kf   = (float*) carve((size_t)NTOK * DD * 4);
    float*  vf   = (float*) carve((size_t)NTOK * DD * 4);
    bf16_t* obf  = (bf16_t*)carve((size_t)NTOK * DD * 2);
    float*  tmp  = (float*) carve((size_t)NTOK * DD * 4);      // h2 / o2 / f2
    bf16_t* big  = (bf16_t*)carve((size_t)NTOK * FFD * 2);     // Xgather / FFN hidden
    bf16_t* h1bf = (bf16_t*)carve((size_t)NTOK * 512 * 2);
    bf16_t* wbf  = (bf16_t*)carve((size_t)NOUT * DD * 2);      // bf16 weight staging
    bf16_t* ybf  = (bf16_t*)carve((size_t)NTOK * DD * 2);

    auto conv = [&](const float* src, bf16_t* dst, int n) {
        f32_to_bf16_kernel<<<(n + 255) / 256, 256, 0, stream>>>(src, dst, n);
    };
    auto gemm = [&](int act, const bf16_t* A, const bf16_t* W, const float* bias,
                    float* Cf, bf16_t* Cb, int M, int N, int K) {
        int tiles  = ((M + 31) / 32) * ((N + 63) / 64);
        int blocks = (tiles + 7) / 8;   // 8 waves / block
        if (act == ACT_LEAKY)
            gemm_bf16_kernel<ACT_LEAKY><<<blocks, 256, 0, stream>>>(A, W, bias, Cf, Cb, M, N, K);
        else if (act == ACT_GELU)
            gemm_bf16_kernel<ACT_GELU><<<blocks, 256, 0, stream>>>(A, W, bias, Cf, Cb, M, N, K);
        else
            gemm_bf16_kernel<ACT_NONE><<<blocks, 256, 0, stream>>>(A, W, bias, Cf, Cb, M, N, K);
    };

    zero_f32_kernel<<<(NTOK * DD + 255) / 256, 256, 0, stream>>>(y, NTOK * DD);

    for (int t = 0; t < 4; ++t) {
        const int ti = 1 + t * 42;
        const int E  = EMB[t];
        const int LH = LHS[t];
        const float* table = (const float*)d_in[ti + 0];
        const float* w1    = (const float*)d_in[ti + 1];
        const float* b1    = (const float*)d_in[ti + 2];
        const float* w2    = (const float*)d_in[ti + 3];
        const float* b2    = (const float*)d_in[ti + 4];
        const float* pos   = (const float*)d_in[ti + 5];
        const float* ln0w  = (const float*)d_in[ti + 6];
        const float* ln0b  = (const float*)d_in[ti + 7];
        const float* topw  = (const float*)d_in[ti + 8];
        const float* topb  = (const float*)d_in[ti + 9];

        // Embedding MLP: gather -> leaky(lin1) -> leaky(lin2)
        gather_bf16_kernel<<<NTOK, 256, 0, stream>>>(x, table, big, E);
        conv(w1, wbf, LH * E);
        gemm(ACT_LEAKY, big, wbf, b1, nullptr, h1bf, NTOK, LH, E);
        conv(w2, wbf, DD * LH);
        gemm(ACT_LEAKY, h1bf, wbf, b2, tmp, nullptr, NTOK, DD, LH);
        ln0_kernel<<<NTOK, DD, 0, stream>>>(tmp, pos, ln0w, ln0b, h, hbf);

        for (int l = 0; l < 2; ++l) {
            const int base = ti + 10 + 16 * l;
            const float* wq  = (const float*)d_in[base + 0];
            const float* bq  = (const float*)d_in[base + 1];
            const float* wk  = (const float*)d_in[base + 2];
            const float* bk  = (const float*)d_in[base + 3];
            const float* wv  = (const float*)d_in[base + 4];
            const float* bv  = (const float*)d_in[base + 5];
            const float* wo  = (const float*)d_in[base + 6];
            const float* bo  = (const float*)d_in[base + 7];
            const float* a1  = (const float*)d_in[base + 8];
            const float* c1  = (const float*)d_in[base + 9];
            const float* fw1 = (const float*)d_in[base + 10];
            const float* fb1 = (const float*)d_in[base + 11];
            const float* fw2 = (const float*)d_in[base + 12];
            const float* fb2 = (const float*)d_in[base + 13];
            const float* a2  = (const float*)d_in[base + 14];
            const float* c2  = (const float*)d_in[base + 15];

            conv(wq, wbf, DD * DD);
            gemm(ACT_NONE, hbf, wbf, bq, qf, nullptr, NTOK, DD, DD);
            conv(wk, wbf, DD * DD);
            gemm(ACT_NONE, hbf, wbf, bk, kf, nullptr, NTOK, DD, DD);
            conv(wv, wbf, DD * DD);
            gemm(ACT_NONE, hbf, wbf, bv, vf, nullptr, NTOK, DD, DD);

            attn_kernel<<<BB * HH * LLEN, 256, 0, stream>>>(qf, kf, vf, x, obf);

            conv(wo, wbf, DD * DD);
            gemm(ACT_NONE, obf, wbf, bo, tmp, nullptr, NTOK, DD, DD);
            cln_res_kernel<<<NTOK, DD, 0, stream>>>(h, tmp, a1, c1, hbf);

            conv(fw1, wbf, FFD * DD);
            gemm(ACT_GELU, hbf, wbf, fb1, nullptr, big, NTOK, FFD, DD);
            conv(fw2, wbf, DD * FFD);
            gemm(ACT_NONE, big, wbf, fb2, tmp, nullptr, NTOK, DD, FFD);
            cln_res_kernel<<<NTOK, DD, 0, stream>>>(h, tmp, a2, c2, hbf);
        }

        topln_accum_kernel<<<NTOK, DD, 0, stream>>>(h, topw, topb, y);
    }

    // -------- vocab projection: d_out = y @ out_w^T + out_b --------
    f32_to_bf16_kernel<<<(NTOK * DD + 255) / 256, 256, 0, stream>>>(y, ybf, NTOK * DD);
    conv(out_w, wbf, NOUT * DD);
    gemm(ACT_NONE, ybf, wbf, out_b, (float*)d_out, nullptr, NTOK, NOUT, DD);
}
